// VSSBlock_90271622627930
// MI455X (gfx1250) — matrix operand order repaired and checked by reading, not compiled, not run on hardware
//
#include <hip/hip_runtime.h>
#include <hip/hip_bf16.h>

// ---------------- problem constants (match reference) ----------------
#define B_SZ   8
#define L_SZ   1024
#define DMODEL 512
#define DINNER 1024      // EXPAND * D_MODEL
#define DSTATE 16
#define DTRANK 32
#define NROWS  (B_SZ * L_SZ)   // 8192 token rows

typedef __bf16 bf16;
typedef __attribute__((ext_vector_type(16))) __bf16 v16bf;
typedef __attribute__((ext_vector_type(8)))  __bf16 v8bf;
typedef __attribute__((ext_vector_type(8)))  float  v8f;
typedef __attribute__((ext_vector_type(4)))  unsigned int v4u;
typedef __attribute__((ext_vector_type(8)))  int v8i;
typedef __attribute__((ext_vector_type(4)))  int v4i;

__device__ __forceinline__ float silu_f(float x) { return x * (1.0f / (1.0f + __expf(-x))); }
__device__ __forceinline__ float softplus_f(float x) { return (x > 20.0f) ? x : log1pf(__expf(x)); }

// ---------------- Tensor Data Mover: 2D tile -> LDS ----------------
// D# per CDNA5 ISA.
//   group0: count=1 | lds_addr | global_addr[56:0] | type=2
//   group1: wg_mask=0, data_size, pad ctl, tensor dims (== tile dims; tiles
//           always in-bounds here), tile dims, tensor_dim0_stride
//   groups 2/3 (+ trailing group): zero (2D tensor, tile_dim2=0)
// NOTE: this toolchain's builtin is the 6-arg form:
//   (uint32x4, int32x8, int32x4, int32x4, int32x8, i32 cpol)
__device__ __forceinline__ void tdm_load_2d(unsigned lds_off, const void* gptr,
                                            unsigned tile_d0, unsigned tile_d1,
                                            unsigned long long stride0_elems,
                                            unsigned dsz_code,          // 0:1B 1:2B 2:4B 3:8B
                                            bool pad_en, unsigned pad_interval,
                                            unsigned pad_amount) {
  unsigned long long ga = (unsigned long long)(uintptr_t)gptr;
  v4u g0;
  g0.x = 1u;                                                // count=1, user descriptor
  g0.y = lds_off;                                           // LDS byte address
  g0.z = (unsigned)(ga & 0xFFFFFFFFu);                      // global_addr[31:0]
  g0.w = (unsigned)((ga >> 32) & 0x1FFFFFFu) | (2u << 30);  // global_addr[56:32] | type=2

  unsigned td0 = tile_d0, td1 = tile_d1;                    // tensor dims == tile dims
  v8i g1;
  g1[0] = (int)((dsz_code << 16) |
                (pad_en ? ((1u << 20) | (pad_interval << 22) | (pad_amount << 25)) : 0u));
  g1[1] = (int)((td0 & 0xFFFFu) << 16);                     // tensor_dim0[15:0] @ bits63:48
  g1[2] = (int)(((td0 >> 16) & 0xFFFFu) | ((td1 & 0xFFFFu) << 16));
  g1[3] = (int)(((td1 >> 16) & 0xFFFFu) | ((tile_d0 & 0xFFFFu) << 16)); // tile_dim0 @ bits127:112
  g1[4] = (int)(tile_d1 & 0xFFFFu);                         // tile_dim1 @ bits143:128, tile_dim2=0
  g1[5] = (int)(stride0_elems & 0xFFFFFFFFull);             // tensor_dim0_stride[31:0]
  g1[6] = (int)((stride0_elems >> 32) & 0xFFFFull);         // stride[47:32]; dim1_stride unused
  g1[7] = 0;
  v4i z4 = {0, 0, 0, 0};
  v8i z8 = {0, 0, 0, 0, 0, 0, 0, 0};
  __builtin_amdgcn_tensor_load_to_lds(g0, g1, z4, z4, z8, 0);
}

// ---------------- f32 -> bf16 elementwise convert ----------------
__global__ void cvt_bf16_kernel(const float* __restrict__ in, bf16* __restrict__ out, int n) {
  int i = blockIdx.x * blockDim.x + threadIdx.x;
  if (i < n) out[i] = (bf16)in[i];
}

// ---------------- LayerNorm over D_MODEL, emit bf16 ----------------
__global__ void layernorm_kernel(const float* __restrict__ x, const float* __restrict__ w,
                                 const float* __restrict__ b, bf16* __restrict__ xn) {
  int row  = blockIdx.x * 8 + (threadIdx.x >> 5);
  int lane = threadIdx.x & 31;
  const float* xr = x + (size_t)row * DMODEL;
  float v[16];
  float s = 0.f, s2 = 0.f;
#pragma unroll
  for (int i = 0; i < 16; ++i) {
    v[i] = xr[lane + i * 32];
    s  += v[i];
    s2 += v[i] * v[i];
  }
#pragma unroll
  for (int off = 16; off > 0; off >>= 1) {
    s  += __shfl_xor(s,  off);
    s2 += __shfl_xor(s2, off);
  }
  float mu  = s  * (1.0f / DMODEL);
  float var = s2 * (1.0f / DMODEL) - mu * mu;
  float inv = rsqrtf(var + 1e-5f);
#pragma unroll
  for (int i = 0; i < 16; ++i) {
    int c = lane + i * 32;
    xn[(size_t)row * DMODEL + c] = (bf16)((v[i] - mu) * inv * w[c] + b[c]);
  }
}

// ---------------- bf16 WMMA GEMM with TDM-staged B tiles ----------------
// C(M,N) = A(M,K) * Bt(N,K)^T. Block = 4 waves M-stacked over one 16*NT-column
// stripe, so the B tile is shared block-wide: TDM double-buffers it in LDS
// (padded 4 DWORDs / 64B -> 80B column stride: bank-conflict-free b128 reads).
// Per-wave tile: (16*MT) x (16*NT).
#define BCOL_STRIDE 40   // bf16 elements per staged column: 32 data + 8 pad (80B)

template<int MT, int NT, bool RES, typename OUT>
__global__ void gemm_wmma_kernel(const bf16* __restrict__ A, const bf16* __restrict__ Bt,
                                 OUT* __restrict__ C, const float* __restrict__ res,
                                 int M, int N, int K) {
  const int tid     = threadIdx.x;
  const int lane    = tid & 31;
  const int wv      = tid >> 5;         // 0..3, M-stacked waves
  const int half_id = lane >> 4;        // 0: lanes 0-15, 1: lanes 16-31
  const int lr      = lane & 15;

  const int nbn  = N / (16 * NT);
  const int bm   = blockIdx.x / nbn;
  const int bn   = blockIdx.x - bm * nbn;
  const int row0 = (bm * 4 + wv) * (16 * MT);
  const int col0 = bn * (16 * NT);

  __shared__ __align__(16) bf16 sB[2][(16 * NT) * BCOL_STRIDE];

  v8f acc[MT][NT] = {};
  const int nk = K / 32;

  // preload k-tile 0 into buffer 0 (TDM, wave 0 of block)
  if (tid < 32) {
    tdm_load_2d((unsigned)(uintptr_t)&sB[0][0], Bt + (size_t)col0 * K,
                /*tile_d0=*/32, /*tile_d1=*/16 * NT, (unsigned long long)K,
                /*dsz=*/1, /*pad=*/true, /*interval 64B=*/3, /*amount 4 DW=*/3);
    __builtin_amdgcn_s_wait_tensorcnt(0);
  }
  __syncthreads();

  for (int kk = 0; kk < nk; ++kk) {
    const int buf = kk & 1;
    const int k0  = kk * 32;
    if (kk + 1 < nk && tid < 32)
      tdm_load_2d((unsigned)(uintptr_t)&sB[buf ^ 1][0],
                  Bt + (size_t)col0 * K + (k0 + 32),
                  32, 16 * NT, (unsigned long long)K, 1, true, 3, 3);

    // A fragments from global: lane lr = row; half 0 -> K {0..7,16..23}, half 1 -> {8..15,24..31}
    v16bf af[MT];
#pragma unroll
    for (int mi = 0; mi < MT; ++mi) {
      const bf16* p = A + (size_t)(row0 + mi * 16 + lr) * K + k0 + half_id * 8;
      v8bf lo = *(const v8bf*)p;
      v8bf hi = *(const v8bf*)(p + 16);
      af[mi] = __builtin_shufflevector(lo, hi, 0,1,2,3,4,5,6,7,8,9,10,11,12,13,14,15);
    }
    // B fragments from LDS: lane lr = column; half 0 -> K {0..7,16..23}, half 1 -> {8..15,24..31}
    v16bf bfr[NT];
#pragma unroll
    for (int ni = 0; ni < NT; ++ni) {
      const bf16* q = &sB[buf][(ni * 16 + lr) * BCOL_STRIDE + half_id * 8];
      v8bf lo = *(const v8bf*)q;
      v8bf hi = *(const v8bf*)(q + 16);
      bfr[ni] = __builtin_shufflevector(lo, hi, 0,1,2,3,4,5,6,7,8,9,10,11,12,13,14,15);
    }
#pragma unroll
    for (int mi = 0; mi < MT; ++mi)
#pragma unroll
      for (int ni = 0; ni < NT; ++ni)
        acc[mi][ni] = __builtin_amdgcn_wmma_f32_16x16x32_bf16(
            false, af[mi], false, bfr[ni], (short)0, acc[mi][ni], false, false);

    if (tid < 32) __builtin_amdgcn_s_wait_tensorcnt(0);  // next buffer landed
    __syncthreads();                                     // all waves done with buf
  }

  // D layout: VGPR v -> row v + 8*half_id, col = lane&15
#pragma unroll
  for (int mi = 0; mi < MT; ++mi)
#pragma unroll
    for (int ni = 0; ni < NT; ++ni)
#pragma unroll
      for (int v = 0; v < 8; ++v) {
        int row = row0 + mi * 16 + half_id * 8 + v;
        int col = col0 + ni * 16 + lr;
        float val = acc[mi][ni][v];
        if constexpr (RES) val += res[(size_t)row * N + col];
        C[(size_t)row * N + col] = (OUT)val;
      }
}

// ---------------- causal depthwise conv (width 4) + SiLU, emit bf16 ----------------
__global__ void dwconv_kernel(const bf16* __restrict__ xz, const float* __restrict__ cw,
                              const float* __restrict__ cb, bf16* __restrict__ u) {
  int d = blockIdx.x * blockDim.x + threadIdx.x;   // 0..DINNER-1
  int t = blockIdx.y;
  int b = blockIdx.z;
  float acc = cb[d];
#pragma unroll
  for (int j = 0; j < 4; ++j) {
    int ts = t - 3 + j;
    if (ts >= 0)
      acc += cw[d * 4 + j] * (float)xz[(size_t)(b * L_SZ + ts) * (2 * DINNER) + d];
  }
  u[(size_t)(b * L_SZ + t) * DINNER + d] = (bf16)silu_f(acc);
}

// ---------------- dt = softplus(dt_r @ W_dt^T + dt_bias) ----------------
__global__ void dt_kernel(const float* __restrict__ xdbl, const float* __restrict__ Wdt,
                          const float* __restrict__ dt_bias, float* __restrict__ dt) {
  int row = blockIdx.y;
  int d   = blockIdx.x * blockDim.x + threadIdx.x;
  __shared__ float sx[DTRANK];
  if (threadIdx.x < DTRANK) sx[threadIdx.x] = xdbl[(size_t)row * 64 + threadIdx.x];
  __syncthreads();
  float acc = dt_bias[d];
#pragma unroll
  for (int r = 0; r < DTRANK; ++r) acc += sx[r] * Wdt[(size_t)d * DTRANK + r];
  dt[(size_t)row * DINNER + d] = softplus_f(acc);
}

// ---------------- selective scan + skip + SiLU(z) gate, emit bf16 ----------------
// one thread per (batch, channel); h[16] in registers; B_t/C_t chunks (64
// timesteps x 32 f32, row stride 64 f32) staged via TDM 2D tile loads.
__global__ void scan_kernel(const float* __restrict__ dt, const bf16* __restrict__ u,
                            const float* __restrict__ xdbl, const bf16* __restrict__ xz,
                            const float* __restrict__ A_log, const float* __restrict__ D_skip,
                            bf16* __restrict__ yout) {
  int b = blockIdx.y;
  int d = blockIdx.x * blockDim.x + threadIdx.x;

  float A_r[DSTATE];
#pragma unroll
  for (int s = 0; s < DSTATE; ++s) A_r[s] = -__expf(A_log[(size_t)d * DSTATE + s]);
  float Dsk = D_skip[d];
  float h[DSTATE] = {};

  __shared__ __align__(16) float sBC[64][32];   // [t_local][0..15]=B, [16..31]=C

  for (int t0 = 0; t0 < L_SZ; t0 += 64) {
    __syncthreads();                            // previous chunk fully consumed
    if (threadIdx.x < 32) {
      tdm_load_2d((unsigned)(uintptr_t)&sBC[0][0],
                  xdbl + (size_t)(b * L_SZ + t0) * 64 + 32,
                  /*tile_d0=*/32, /*tile_d1=*/64, /*stride0=*/64ull,
                  /*dsz=*/2, /*pad=*/false, 0, 0);
      __builtin_amdgcn_s_wait_tensorcnt(0);
    }
    __syncthreads();

    for (int tl = 0; tl < 64; ++tl) {
      size_t rowi = (size_t)(b * L_SZ + t0 + tl);
      float dtv = dt[rowi * DINNER + d];
      float uv  = (float)u[rowi * DINNER + d];
      float zv  = (float)xz[rowi * (2 * DINNER) + DINNER + d];
      float du  = dtv * uv;
      float y   = 0.f;
#pragma unroll
      for (int s = 0; s < DSTATE; ++s) {
        float e = __expf(dtv * A_r[s]);
        h[s] = e * h[s] + du * sBC[tl][s];
        y += h[s] * sBC[tl][16 + s];
      }
      float yf = (y + uv * Dsk) * silu_f(zv);
      yout[rowi * DINNER + d] = (bf16)yf;
    }
  }
}

// ---------------- host-side GEMM launcher ----------------
template<int MT, int NT, bool RES, typename OUT>
static void launch_gemm(const bf16* A, const bf16* Bt, OUT* C, const float* res,
                        int M, int N, int K, hipStream_t stream) {
  int nbm = M / (16 * MT * 4);   // 4 waves M-stacked per block
  int nbn = N / (16 * NT);
  gemm_wmma_kernel<MT, NT, RES, OUT><<<nbm * nbn, 128, 0, stream>>>(A, Bt, C, res, M, N, K);
}

extern "C" void kernel_launch(void* const* d_in, const int* in_sizes, int n_in,
                              void* d_out, int out_size, void* d_ws, size_t ws_size,
                              hipStream_t stream) {
  (void)in_sizes; (void)n_in; (void)out_size; (void)ws_size;

  const float* x       = (const float*)d_in[0];   // (B, L, DMODEL)
  const float* ln_w    = (const float*)d_in[1];
  const float* ln_b    = (const float*)d_in[2];
  const float* W_in    = (const float*)d_in[3];   // (2*DINNER, DMODEL)
  const float* conv_w  = (const float*)d_in[4];   // (DINNER, 1, 4)
  const float* conv_b  = (const float*)d_in[5];
  const float* W_xproj = (const float*)d_in[6];   // (64, DINNER)
  const float* W_dt    = (const float*)d_in[7];   // (DINNER, DTRANK)
  const float* dt_bias = (const float*)d_in[8];
  const float* A_log   = (const float*)d_in[9];   // (DINNER, DSTATE)
  const float* D_skip  = (const float*)d_in[10];
  const float* W_out   = (const float*)d_in[11];  // (DMODEL, DINNER)

  // ---- workspace carve-up (256B aligned chunks) ----
  size_t off = 0;
  char* ws = (char*)d_ws;
  auto take = [&](size_t bytes) -> void* {
    void* p = ws + off;
    off += (bytes + 255) & ~(size_t)255;
    return p;
  };
  bf16*  W_in_bf  = (bf16*) take((size_t)(2 * DINNER) * DMODEL * 2);
  bf16*  W_xp_bf  = (bf16*) take((size_t)(DTRANK + 2 * DSTATE) * DINNER * 2);
  bf16*  W_out_bf = (bf16*) take((size_t)DMODEL * DINNER * 2);
  bf16*  xn_bf    = (bf16*) take((size_t)NROWS * DMODEL * 2);
  bf16*  xz_bf    = (bf16*) take((size_t)NROWS * (2 * DINNER) * 2);
  bf16*  u_bf     = (bf16*) take((size_t)NROWS * DINNER * 2);
  float* xdbl     = (float*)take((size_t)NROWS * 64 * 4);
  float* dtb      = (float*)take((size_t)NROWS * DINNER * 4);
  bf16*  y_bf     = (bf16*) take((size_t)NROWS * DINNER * 2);

  // ---- 1. weight conversions to bf16 ----
  {
    int n1 = 2 * DINNER * DMODEL;
    cvt_bf16_kernel<<<(n1 + 255) / 256, 256, 0, stream>>>(W_in, W_in_bf, n1);
    int n2 = (DTRANK + 2 * DSTATE) * DINNER;
    cvt_bf16_kernel<<<(n2 + 255) / 256, 256, 0, stream>>>(W_xproj, W_xp_bf, n2);
    int n3 = DMODEL * DINNER;
    cvt_bf16_kernel<<<(n3 + 255) / 256, 256, 0, stream>>>(W_out, W_out_bf, n3);
  }

  // ---- 2. LayerNorm -> xn (bf16) ----
  layernorm_kernel<<<NROWS / 8, 256, 0, stream>>>(x, ln_w, ln_b, xn_bf);

  // ---- 3. in_proj GEMM: xz = xn @ W_in^T  (8192 x 2048 x 512), bf16 out ----
  launch_gemm<2, 4, false, bf16>(xn_bf, W_in_bf, xz_bf, nullptr,
                                 NROWS, 2 * DINNER, DMODEL, stream);

  // ---- 4. depthwise causal conv + SiLU -> u (bf16) ----
  dwconv_kernel<<<dim3(DINNER / 256, L_SZ, B_SZ), 256, 0, stream>>>(xz_bf, conv_w, conv_b, u_bf);

  // ---- 5. x_proj GEMM: x_dbl = u @ W_xproj^T  (8192 x 64 x 1024), f32 out ----
  launch_gemm<2, 4, false, float>(u_bf, W_xp_bf, xdbl, nullptr,
                                  NROWS, DTRANK + 2 * DSTATE, DINNER, stream);

  // ---- 6. dt = softplus(dt_r @ W_dt^T + dt_bias), f32 ----
  dt_kernel<<<dim3(DINNER / 256, NROWS), 256, 0, stream>>>(xdbl, W_dt, dt_bias, dtb);

  // ---- 7. selective scan + D-skip + SiLU(z) gate -> y (bf16) ----
  scan_kernel<<<dim3(DINNER / 256, B_SZ), 256, 0, stream>>>(dtb, u_bf, xdbl, xz_bf,
                                                            A_log, D_skip, y_bf);

  // ---- 8. out_proj GEMM + residual: out = x + y @ W_out^T  (8192 x 512 x 1024) ----
  launch_gemm<2, 4, true, float>(y_bf, W_out_bf, (float*)d_out, x,
                                 NROWS, DMODEL, DINNER, stream);
}
